// SelfAttention_13993003450718
// MI455X (gfx1250) — compile-verified, matrix operand order checked by reading, not run on hardware
//
#include <hip/hip_runtime.h>
#include <hip/hip_bf16.h>
#include <stdint.h>

typedef _Float16 f16;
typedef _Float16 v16h __attribute__((ext_vector_type(16)));
typedef _Float16 v8h  __attribute__((ext_vector_type(8)));
typedef _Float16 v4h  __attribute__((ext_vector_type(4)));
typedef float    v8f  __attribute__((ext_vector_type(8)));

#define N_TOK 8192
#define DDIM  1024
#define RDIM  1024
#define BK    128     // keys per flash iteration
#define LDA   1040    // LDS row stride (halves) for Q tile (padded vs 1024)
#define LDS_S 136     // LDS row stride for S (f32) / P (f16) tiles (padded vs 128)

// NT fragment loader for v_wmma_f32_16x16x32_f16.
// `base` points at element [0][kbase] of a row-major tile with row stride `ld` (halves).
//   lane l (0..15):  row l,   K = {0..7, 16..23}
//   lane l+16:       row l,   K = {8..15, 24..31}
__device__ __forceinline__ v16h load_frag(const f16* base, int ld, int lane) {
  const int m = lane & 15;
  const int h = (lane >> 4) & 1;
  const f16* p = base + (size_t)m * ld + h * 8;
  v8h lo = *(const v8h*)(p);
  v8h hi = *(const v8h*)(p + 16);
  return __builtin_shufflevector(lo, hi, 0,1,2,3,4,5,6,7,8,9,10,11,12,13,14,15);
}

__device__ __forceinline__ v8f wmma_f16(v16h a, v16h b, v8f c) {
  return __builtin_amdgcn_wmma_f32_16x16x32_f16(false, a, false, b, (short)0, c,
                                                false, false);
}

// monotone float<->uint mapping so ds_max_u32 implements float max
__device__ __forceinline__ unsigned f2mono(float f) {
  unsigned u = __float_as_uint(f);
  return (u & 0x80000000u) ? ~u : (u | 0x80000000u);
}
__device__ __forceinline__ float mono2f(unsigned u) {
  return (u & 0x80000000u) ? __uint_as_float(u & 0x7fffffffu)
                           : __uint_as_float(~u);
}

// ---------------------------------------------------------------- convert f32 -> f16
__global__ void __launch_bounds__(256)
cvt_f32_f16(const float* __restrict__ in, f16* __restrict__ out, int n4) {
  int i = blockIdx.x * blockDim.x + threadIdx.x;
  if (i >= n4) return;
  float4 v = ((const float4*)in)[i];
  v4h o;
  o.x = (f16)v.x; o.y = (f16)v.y; o.z = (f16)v.z; o.w = (f16)v.w;
  ((v4h*)out)[i] = o;
}

// ---------------------------------------------------------------- projection GEMM (NT)
// out = X[N,D] @ W[R,D]^T + bias, written as f16.
// transpose_out==0: out is [N,R] row-major. transpose_out==1: out is [R,N] row-major.
// Block: 256 threads = 8 waves; wave w -> rows blockIdx.x*256 + w*32 (two 16-row tiles),
// cols blockIdx.y*64 (four 16-col tiles). Software-pipelined over the D reduction.
__global__ void __launch_bounds__(256)
proj_gemm(const f16* __restrict__ X, const f16* __restrict__ W,
          const float* __restrict__ bias, f16* __restrict__ out,
          int transpose_out)
{
  const int lane = threadIdx.x & 31;
  const int wave = threadIdx.x >> 5;
  const int row0 = blockIdx.x * 256 + wave * 32;
  const int col0 = blockIdx.y * 64;

  const f16* xa0 = X + (size_t)row0 * DDIM;
  const f16* xa1 = X + (size_t)(row0 + 16) * DDIM;
  const f16* wb  = W + (size_t)col0 * DDIM;

  v8f acc[2][4] = {};

  // prologue: fragments for d = 0
  v16h a0 = load_frag(xa0, DDIM, lane);
  v16h a1 = load_frag(xa1, DDIM, lane);
  v16h b0 = load_frag(wb + 0 * 16 * DDIM, DDIM, lane);
  v16h b1 = load_frag(wb + 1 * 16 * DDIM, DDIM, lane);
  v16h b2 = load_frag(wb + 2 * 16 * DDIM, DDIM, lane);
  v16h b3 = load_frag(wb + 3 * 16 * DDIM, DDIM, lane);

  for (int d = 0; d < DDIM - 32; d += 32) {
    const int dn = d + 32;
    // prefetch next reduction chunk while current WMMAs execute
    v16h a0n = load_frag(xa0 + dn, DDIM, lane);
    v16h a1n = load_frag(xa1 + dn, DDIM, lane);
    v16h b0n = load_frag(wb + 0 * 16 * DDIM + dn, DDIM, lane);
    v16h b1n = load_frag(wb + 1 * 16 * DDIM + dn, DDIM, lane);
    v16h b2n = load_frag(wb + 2 * 16 * DDIM + dn, DDIM, lane);
    v16h b3n = load_frag(wb + 3 * 16 * DDIM + dn, DDIM, lane);

    acc[0][0] = wmma_f16(a0, b0, acc[0][0]);
    acc[0][1] = wmma_f16(a0, b1, acc[0][1]);
    acc[0][2] = wmma_f16(a0, b2, acc[0][2]);
    acc[0][3] = wmma_f16(a0, b3, acc[0][3]);
    acc[1][0] = wmma_f16(a1, b0, acc[1][0]);
    acc[1][1] = wmma_f16(a1, b1, acc[1][1]);
    acc[1][2] = wmma_f16(a1, b2, acc[1][2]);
    acc[1][3] = wmma_f16(a1, b3, acc[1][3]);

    a0 = a0n; a1 = a1n; b0 = b0n; b1 = b1n; b2 = b2n; b3 = b3n;
  }
  acc[0][0] = wmma_f16(a0, b0, acc[0][0]);
  acc[0][1] = wmma_f16(a0, b1, acc[0][1]);
  acc[0][2] = wmma_f16(a0, b2, acc[0][2]);
  acc[0][3] = wmma_f16(a0, b3, acc[0][3]);
  acc[1][0] = wmma_f16(a1, b0, acc[1][0]);
  acc[1][1] = wmma_f16(a1, b1, acc[1][1]);
  acc[1][2] = wmma_f16(a1, b2, acc[1][2]);
  acc[1][3] = wmma_f16(a1, b3, acc[1][3]);

  // C layout: VGPR v, lanes 0-15 -> row v; lanes 16-31 -> row v+8; col = lane&15
  const int m = lane & 15;
  const int h = (lane >> 4) & 1;
#pragma unroll
  for (int r = 0; r < 2; ++r) {
#pragma unroll
    for (int ct = 0; ct < 4; ++ct) {
      const int col = col0 + ct * 16 + m;
      const float bv = bias[col];
      const int rbase = row0 + r * 16;
      if (transpose_out) {
        v8h pk;
#pragma unroll
        for (int v = 0; v < 8; ++v) pk[v] = (f16)(acc[r][ct][v] + bv);
        *(v8h*)(out + (size_t)col * N_TOK + rbase + 8 * h) = pk;
      } else {
#pragma unroll
        for (int v = 0; v < 8; ++v)
          out[(size_t)(rbase + v + 8 * h) * RDIM + col] = (f16)(acc[r][ct][v] + bv);
      }
    }
  }
}

// ---------------------------------------------------------------- flash attention
// One block (8 waves / 256 threads) per 16 query rows. Online softmax over key blocks
// of BK=128. Wave w computes score subtile j = w*16..w*16+15 and owns output columns
// w*128..w*128+127 (16x128 f32 accumulator = 64 VGPRs/lane).
__global__ void __launch_bounds__(256)
flash_attn(const f16* __restrict__ Qh, const f16* __restrict__ Kh,
           const f16* __restrict__ Vt, float* __restrict__ out)
{
  __shared__ __align__(16) f16      Qsh[16 * LDA];
  __shared__ __align__(16) float    Ssh[16 * LDS_S];
  __shared__ __align__(16) f16      Psh[16 * LDS_S];
  __shared__ float    msh[16], lsh[16], rsh[16], rowsum[16];
  __shared__ unsigned bmax[16];

  const int tid  = threadIdx.x;
  const int lane = tid & 31;
  const int wave = tid >> 5;
  const int q0   = blockIdx.x * 16;

  // stage Q tile (16 x 1024 halves) in LDS with padded stride
  for (int c = tid; c < 16 * (DDIM / 8); c += 256) {
    int row = c >> 7;            // 128 8-half chunks per row
    int off = (c & 127) * 8;
    *(v8h*)(&Qsh[row * LDA + off]) =
        *(const v8h*)(Qh + (size_t)(q0 + row) * DDIM + off);
  }
  if (tid < 16) {
    msh[tid] = -3.0e38f; lsh[tid] = 0.0f; bmax[tid] = 0u;
  }
  __syncthreads();

  v8f acc[8] = {};
  const float alpha = 0.045084294f;  // (1/sqrt(1024)) * log2(e)
  const int   cb    = wave * 128;    // this wave's output-column base

  for (int J = 0; J < N_TOK; J += BK) {
    // ---- scores: S[0:16, wave*16 : wave*16+16] = Q @ K_J^T  (pipelined over D)
    {
      const f16* kbase = Kh + (size_t)(J + wave * 16) * DDIM;
      if (J + BK < N_TOK) {
        // hint the next key block toward the caches (global_prefetch_b8)
        __builtin_prefetch(Kh + (size_t)(J + BK + wave * 16) * DDIM + lane * 64, 0, 1);
      }
      v8f s = {};
      v16h a = load_frag(Qsh, LDA, lane);
      v16h b = load_frag(kbase, DDIM, lane);
      for (int d = 0; d < DDIM - 32; d += 32) {
        v16h an = load_frag(Qsh + d + 32, LDA, lane);
        v16h bn = load_frag(kbase + d + 32, DDIM, lane);
        s = wmma_f16(a, b, s);
        a = an; b = bn;
      }
      s = wmma_f16(a, b, s);

      const int m = lane & 15, h = (lane >> 4) & 1;
#pragma unroll
      for (int v = 0; v < 8; ++v)
        Ssh[(v + 8 * h) * LDS_S + wave * 16 + m] = s[v];
    }
    __syncthreads();

    // ---- each thread owns 8 contiguous score elements of one row
    const int e0  = tid * 8;
    const int row = e0 >> 7;
    const int col = e0 & 127;
    float sv[8];
    {
      float lm = -3.0e38f;
#pragma unroll
      for (int j = 0; j < 8; ++j) {
        sv[j] = Ssh[row * LDS_S + col + j] * alpha;
        lm = fmaxf(lm, sv[j]);
      }
      atomicMax(&bmax[row], f2mono(lm));   // ds_max_u32
    }
    __syncthreads();

    // ---- combine running max, rescale factor
    if (tid < 16) {
      float mo = msh[tid];
      float mx = fmaxf(mo, mono2f(bmax[tid]));
      msh[tid]    = mx;
      rsh[tid]    = __builtin_amdgcn_exp2f(mo - mx);
      rowsum[tid] = 0.0f;
      bmax[tid]   = 0u;                    // reset for next block
    }
    __syncthreads();

    // ---- probabilities from registers: P = exp2(sv - m) as f16, plus row sums
    {
      const float mrow = msh[row];
      float lsum = 0.0f;
#pragma unroll
      for (int j = 0; j < 8; ++j) {
        float p = __builtin_amdgcn_exp2f(sv[j] - mrow);
        lsum += p;
        Psh[row * LDS_S + col + j] = (f16)p;
      }
      atomicAdd(&rowsum[row], lsum);       // ds_add_f32
    }
    __syncthreads();

    if (tid < 16) lsh[tid] = lsh[tid] * rsh[tid] + rowsum[tid];

    // ---- rescale accumulator, then acc += P @ Vt^T (pipelined over 32 B-fragments)
    {
      const int h = (lane >> 4) & 1;
      float rv[8];
#pragma unroll
      for (int v = 0; v < 8; ++v) rv[v] = rsh[v + 8 * h];
#pragma unroll
      for (int ct = 0; ct < 8; ++ct)
#pragma unroll
        for (int v = 0; v < 8; ++v) acc[ct][v] *= rv[v];

      v16h afrag[4];
#pragma unroll
      for (int kc = 0; kc < 4; ++kc)
        afrag[kc] = load_frag(Psh + kc * 32, LDS_S, lane);

      // flattened (kc, ct) loop, B fragment double-buffered
      v16h bc = load_frag(Vt + (size_t)cb * N_TOK + J, N_TOK, lane);
#pragma unroll
      for (int i = 0; i < 32; ++i) {
        v16h bn = bc;
        if (i < 31) {
          const int kcn = (i + 1) >> 3, ctn = (i + 1) & 7;
          bn = load_frag(Vt + (size_t)(cb + ctn * 16) * N_TOK + J + kcn * 32,
                         N_TOK, lane);
        }
        const int kc = i >> 3, ct = i & 7;
        acc[ct] = wmma_f16(afrag[kc], bc, acc[ct]);
        bc = bn;
      }
    }
    __syncthreads();
  }

  // ---- epilogue: divide by l and store f32 output
  const int m = lane & 15, h = (lane >> 4) & 1;
  float inv[8];
#pragma unroll
  for (int v = 0; v < 8; ++v) inv[v] = 1.0f / lsh[v + 8 * h];
#pragma unroll
  for (int ct = 0; ct < 8; ++ct)
#pragma unroll
    for (int v = 0; v < 8; ++v)
      out[(size_t)(q0 + v + 8 * h) * RDIM + cb + ct * 16 + m] = acc[ct][v] * inv[v];
}

// ---------------------------------------------------------------- launch
extern "C" void kernel_launch(void* const* d_in, const int* in_sizes, int n_in,
                              void* d_out, int out_size, void* d_ws, size_t ws_size,
                              hipStream_t stream) {
  const float* x  = (const float*)d_in[0];
  const float* Wq = (const float*)d_in[1];
  const float* bq = (const float*)d_in[2];
  const float* Wk = (const float*)d_in[3];
  const float* bk = (const float*)d_in[4];
  const float* Wv = (const float*)d_in[5];
  const float* bv = (const float*)d_in[6];
  float* out = (float*)d_out;

  uint8_t* ws = (uint8_t*)d_ws;
  const size_t XH  = (size_t)N_TOK * DDIM * sizeof(f16);   // 16 MB
  const size_t WH  = (size_t)RDIM * DDIM * sizeof(f16);    // 2 MB
  f16* xh  = (f16*)(ws);
  f16* wqh = (f16*)(ws + XH);
  f16* wkh = (f16*)(ws + XH + WH);
  f16* wvh = (f16*)(ws + XH + 2 * WH);
  f16* Qh  = (f16*)(ws + XH + 3 * WH);
  f16* Kh  = (f16*)(ws + XH + 3 * WH + XH);
  f16* Vt  = (f16*)(ws + XH + 3 * WH + 2 * XH);            // total ~73.4 MB

  // f32 -> f16 conversions
  {
    int n4 = (N_TOK * DDIM) / 4;
    cvt_f32_f16<<<n4 / 256, 256, 0, stream>>>(x, xh, n4);
    int w4 = (RDIM * DDIM) / 4;
    cvt_f32_f16<<<w4 / 256, 256, 0, stream>>>(Wq, wqh, w4);
    cvt_f32_f16<<<w4 / 256, 256, 0, stream>>>(Wk, wkh, w4);
    cvt_f32_f16<<<w4 / 256, 256, 0, stream>>>(Wv, wvh, w4);
  }

  // projections (WMMA): Q,K row-major [N,R]; V transposed [R,N]
  dim3 pgrid(N_TOK / 256, RDIM / 64);
  proj_gemm<<<pgrid, 256, 0, stream>>>(xh, wqh, bq, Qh, 0);
  proj_gemm<<<pgrid, 256, 0, stream>>>(xh, wkh, bk, Kh, 0);
  proj_gemm<<<pgrid, 256, 0, stream>>>(xh, wvh, bv, Vt, 1);

  // flash attention (WMMA scores + WMMA PV, online softmax)
  flash_attn<<<N_TOK / 16, 256, 0, stream>>>(Qh, Kh, Vt, out);
}
